// Transformer_50800873177483
// MI455X (gfx1250) — matrix operand order repaired and checked by reading, not compiled, not run on hardware
//
#include <hip/hip_runtime.h>
#include <hip/hip_bf16.h>
#include <math.h>

// ---------------- types ----------------
typedef __bf16 bf16_t;
typedef __attribute__((ext_vector_type(16))) __bf16 v16bf;
typedef __attribute__((ext_vector_type(8)))  float  v8f;
typedef int v4i_t __attribute__((ext_vector_type(4)));

#define VOCAB   32000
#define HID     1024
#define NHEAD   16
#define DK      64
#define FFH     2048
#define BATCH   4
#define SEQ     1000
#define SPAD    1024   // padded seq rows per (b,h)
#define MPAD    4096   // padded token rows
#define MVALID  4000   // B*S

#define GLOBAL_AS __attribute__((address_space(1)))
#define LDS_AS    __attribute__((address_space(3)))

#if defined(__has_builtin)
#if __has_builtin(__builtin_amdgcn_global_load_async_to_lds_b128) && \
    __has_builtin(__builtin_amdgcn_s_wait_asynccnt)
#define HAVE_ASYNC_LDS 1
#endif
#endif

union FragB { v16bf v; uint4 q[2]; };

__device__ __forceinline__ v8f vzero8() {
  v8f z;
#pragma unroll
  for (int i = 0; i < 8; ++i) z[i] = 0.f;
  return z;
}

__device__ __forceinline__ v8f wmma_bf16(v16bf a, v16bf b, v8f c) {
  // D = A(16x32 bf16) x B(32x16 bf16) + C(16x16 f32)
  return __builtin_amdgcn_wmma_f32_16x16x32_bf16(false, a, false, b, (short)0, c,
                                                 false, false);
}

// ---------------- 0) weight prep: fp32 [K,N]-ish -> bf16 transposed [N][K] ---
// in element (k, n) lives at  in[(n>>6)*nstride + k*ldb + (n&63)]
//   normal row-major [K,N]: nstride=64,  ldb=N
//   per-head QKV [H,HID,DK]: nstride=HID*DK, ldb=DK  (n = h*64+dk)
// out[n*K + k] (bf16)
__global__ __launch_bounds__(256) void prep_kernel(const float* __restrict__ in,
                                                   bf16_t* __restrict__ out,
                                                   int K, long nstride, long ldb) {
  __shared__ float tile[64 * 64];   // [k][n]
  int n0 = blockIdx.x * 64;
  int k0 = blockIdx.y * 64;
  int tid = threadIdx.x;
  // phase 1: load 64x64 fp32 tile (rows contiguous in n)
  {
    int k = tid >> 2;
    int n4 = (tid & 3) * 16;
    const float* src = in + (long)(n0 >> 6) * nstride + (long)(k0 + k) * ldb + n4;
#if HAVE_ASYNC_LDS
    float* s = const_cast<float*>(src);
#pragma unroll
    for (int i = 0; i < 4; ++i) {
      __builtin_amdgcn_global_load_async_to_lds_b128(
          (GLOBAL_AS v4i_t*)(s + i * 4),
          (LDS_AS v4i_t*)&tile[k * 64 + n4 + i * 4], 0, 0);
    }
    __builtin_amdgcn_s_wait_asynccnt(0);
#else
    const float4* s4 = reinterpret_cast<const float4*>(src);
#pragma unroll
    for (int i = 0; i < 4; ++i) {
      float4 f = s4[i];
      tile[k * 64 + n4 + i * 4 + 0] = f.x;
      tile[k * 64 + n4 + i * 4 + 1] = f.y;
      tile[k * 64 + n4 + i * 4 + 2] = f.z;
      tile[k * 64 + n4 + i * 4 + 3] = f.w;
    }
#endif
  }
  __syncthreads();
  // phase 2: write transposed bf16, 32B contiguous per thread
  {
    int n = tid & 63;
    int kc = (tid >> 6) * 16;
    union { bf16_t h[8]; uint4 u; } pk;
    bf16_t* orow = out + (long)(n0 + n) * K + (k0 + kc);
#pragma unroll
    for (int g = 0; g < 2; ++g) {
#pragma unroll
      for (int i = 0; i < 8; ++i) pk.h[i] = (bf16_t)tile[(kc + g * 8 + i) * 64 + n];
      reinterpret_cast<uint4*>(orow)[g] = pk.u;
    }
  }
}

// ---------------- 1) embedding + constant positional encoding ----------------
__global__ __launch_bounds__(256) void embed_kernel(const int* __restrict__ tokens,
                                                    const float* __restrict__ emb,
                                                    float* __restrict__ xf,
                                                    bf16_t* __restrict__ xb) {
  int m = blockIdx.x;              // 0..MPAD-1
  int c0 = threadIdx.x * 4;
  if (m >= MVALID) {
#pragma unroll
    for (int u = 0; u < 4; ++u) {
      xf[(long)m * HID + c0 + u] = 0.f;
      xb[(long)m * HID + c0 + u] = (bf16_t)0.f;
    }
    return;
  }
  int tok = tokens[m];
#pragma unroll
  for (int u = 0; u < 4; ++u) {
    int c = c0 + u;
    int i = c & ~1;  // even base index
    float tmp = 999.0f / __powf(10000.0f, (float)i / 1024.0f);
    float pe = (c & 1) ? __cosf(tmp) : __sinf(tmp);
    float val = emb[(long)tok * HID + c] + pe;
    xf[(long)m * HID + c] = val;
    xb[(long)m * HID + c] = (bf16_t)val;
  }
}

// ---------------- 2) WMMA GEMM, A bf16 [4096,K], B bf16 transposed [N][K] ----
// Block: 128 threads = 4 waves; block tile 256 rows x 64 cols.
// Each wave: 4 M-tiles (64 rows) x 64 cols -> 16 WMMAs / K-step, no LDS.
// mode 0: bf16 out token-major.  mode 1: bf16 out [B,H,SPAD,DK] (QKV).
// mode 2: f32 out token-major.
__global__ __launch_bounds__(128) void gemm_kernel(const bf16_t* __restrict__ A,
                                                   const bf16_t* __restrict__ BT,
                                                   const float* __restrict__ bias,
                                                   void* __restrict__ out,
                                                   int N, int K, int mode, int relu) {
  int nt   = blockIdx.x;
  int mb   = blockIdx.y;
  int tid  = threadIdx.x;
  int wave = tid >> 5;
  int lane = tid & 31;
  int hw   = lane >> 4;
  int ln   = lane & 15;
  int mbase = mb * 256 + wave * 64;

  const bf16_t* Arow[4];
  const bf16_t* Brow[4];
#pragma unroll
  for (int mt = 0; mt < 4; ++mt) Arow[mt] = A + (long)(mbase + mt * 16 + ln) * K;
#pragma unroll
  for (int j = 0; j < 4; ++j) Brow[j] = BT + (long)(nt * 64 + j * 16 + ln) * K;

  v8f acc[4][4];
#pragma unroll
  for (int mt = 0; mt < 4; ++mt)
#pragma unroll
    for (int j = 0; j < 4; ++j) acc[mt][j] = vzero8();

  for (int k0 = 0; k0 < K; k0 += 32) {
    if (k0 + 32 < K) {  // emits global_prefetch_b8
      __builtin_prefetch(Brow[0] + k0 + 32);
      __builtin_prefetch(Arow[0] + k0 + 32);
    }
    FragB bfr[4];
#pragma unroll
    for (int j = 0; j < 4; ++j) {
      const uint4* bp = reinterpret_cast<const uint4*>(Brow[j] + k0);
      bfr[j].q[0] = bp[hw + 0];   // K {0..7} | {8..15}
      bfr[j].q[1] = bp[hw + 2];   // K {16..23} | {24..31}
    }
    FragB af[4];
#pragma unroll
    for (int mt = 0; mt < 4; ++mt) {
      const uint4* ap = reinterpret_cast<const uint4*>(Arow[mt] + k0);
      af[mt].q[0] = ap[hw ? 1 : 0];   // K {0..7} | {8..15}
      af[mt].q[1] = ap[hw ? 3 : 2];   // K {16..23} | {24..31}
    }
#pragma unroll
    for (int mt = 0; mt < 4; ++mt)
#pragma unroll
      for (int j = 0; j < 4; ++j)
        acc[mt][j] = wmma_bf16(af[mt].v, bfr[j].v, acc[mt][j]);
  }

  // bias + activation + store (C layout: lanes 0-15 -> M=i, lanes 16-31 -> M=i+8)
#pragma unroll
  for (int j = 0; j < 4; ++j) {
    int col = nt * 64 + j * 16 + ln;
    float bv = bias ? bias[col] : 0.f;
#pragma unroll
    for (int mt = 0; mt < 4; ++mt) {
#pragma unroll
      for (int i = 0; i < 8; ++i) {
        int m = mbase + mt * 16 + hw * 8 + i;
        float val = acc[mt][j][i] + bv;
        if (relu) val = fmaxf(val, 0.f);
        if (m < MVALID) {
          if (mode == 0) {
            ((bf16_t*)out)[(long)m * N + col] = (bf16_t)val;
          } else if (mode == 1) {
            int b = m / SEQ, s = m % SEQ;
            ((bf16_t*)out)[(((long)(b * NHEAD + nt)) * SPAD + s) * DK +
                           (j * 16 + ln)] = (bf16_t)val;
          } else {
            ((float*)out)[(long)m * N + col] = val;
          }
        }
      }
    }
  }
}

// ---------------- 3) flash-style attention ----------------
// scores[s,t] = floor((k_s . q_t) / 8), causal mask t<=s, softmax over t, O = P.V
__global__ __launch_bounds__(256) void attn_kernel(const bf16_t* __restrict__ qg,
                                                   const bf16_t* __restrict__ kg,
                                                   const bf16_t* __restrict__ vg,
                                                   bf16_t* __restrict__ attnb) {
  __shared__ bf16_t vT[64 * 32];        // v tile transposed [dk][t]
  __shared__ bf16_t Pl[8 * 16 * 32];    // per-wave P tile [16 rows][32 t]
  int bh = blockIdx.y;                  // b*16 + h
  int sb = blockIdx.x;                  // 128-row s block
  int tid = threadIdx.x;
  int wave = tid >> 5, lane = tid & 31, hw = lane >> 4, ln = lane & 15;
  int s0w = sb * 128 + wave * 16;
  const bf16_t* kbase = kg + (long)bh * SPAD * DK;
  const bf16_t* qbase = qg + (long)bh * SPAD * DK;
  const bf16_t* vbase = vg + (long)bh * SPAD * DK;

  // A fragments for K rows (constant over the t loop): 2 ksteps covering dk 0..63
  FragB kf0, kf1;
  {
    const uint4* kp = reinterpret_cast<const uint4*>(kbase + (long)(s0w + ln) * DK);
    kf0.q[0] = kp[hw ? 1 : 0];
    kf0.q[1] = kp[hw ? 3 : 2];
    kf1.q[0] = kp[hw ? 5 : 4];
    kf1.q[1] = kp[hw ? 7 : 6];
  }

  v8f O[4];
#pragma unroll
  for (int j = 0; j < 4; ++j) O[j] = vzero8();
  float run_m[8], run_l[8];
#pragma unroll
  for (int i = 0; i < 8; ++i) { run_m[i] = -INFINITY; run_l[i] = 0.f; }

  int tmax = sb * 128 + 127;
  if (tmax > SEQ - 1) tmax = SEQ - 1;
  bf16_t* Pw = Pl + wave * 16 * 32;

  for (int t0 = 0; t0 <= tmax; t0 += 32) {
    // cooperatively stage vT[dk][t] for this 32-column chunk
    {
      int tt  = tid & 31;
      int dk0 = (tid >> 5) * 8;
      const uint4* vp =
          reinterpret_cast<const uint4*>(vbase + (long)(t0 + tt) * DK + dk0);
      uint4 raw = vp[0];
      const bf16_t* ve = reinterpret_cast<const bf16_t*>(&raw);
#pragma unroll
      for (int i = 0; i < 8; ++i) vT[(dk0 + i) * 32 + tt] = ve[i];
    }
    __syncthreads();

    if (t0 <= s0w + 15) {   // wave-uniform: EXEC stays all-ones
      // ---- scores: two 16-col tiles, contraction over dk=64 (2 WMMAs each)
      v8f sc[2];
#pragma unroll
      for (int tt = 0; tt < 2; ++tt) {
        FragB qf0, qf1;
        const uint4* qp =
            reinterpret_cast<const uint4*>(qbase + (long)(t0 + tt * 16 + ln) * DK);
        qf0.q[0] = qp[hw ? 1 : 0];
        qf0.q[1] = qp[hw ? 3 : 2];
        qf1.q[0] = qp[hw ? 5 : 4];
        qf1.q[1] = qp[hw ? 7 : 6];
        v8f a = vzero8();
        a = wmma_bf16(kf0.v, qf0.v, a);
        a = wmma_bf16(kf1.v, qf1.v, a);
        sc[tt] = a;
      }
      // ---- floor-scale + causal mask
#pragma unroll
      for (int tt = 0; tt < 2; ++tt)
#pragma unroll
        for (int i = 0; i < 8; ++i) {
          int s = s0w + hw * 8 + i;
          int t = t0 + tt * 16 + ln;
          float x = floorf(sc[tt][i] * 0.125f);
          sc[tt][i] = (t > s) ? -INFINITY : x;
        }
      // ---- online softmax per row (row lives in one half-wave, one VGPR idx)
#pragma unroll
      for (int i = 0; i < 8; ++i) {
        float mx = fmaxf(sc[0][i], sc[1][i]);
#pragma unroll
        for (int d = 1; d < 16; d <<= 1) mx = fmaxf(mx, __shfl_xor(mx, d, 16));
        float nm = fmaxf(run_m[i], mx);
        float scale = (run_m[i] == -INFINITY) ? 0.f : __expf(run_m[i] - nm);
        float p0 = (sc[0][i] == -INFINITY) ? 0.f : __expf(sc[0][i] - nm);
        float p1 = (sc[1][i] == -INFINITY) ? 0.f : __expf(sc[1][i] - nm);
        float l = p0 + p1;
#pragma unroll
        for (int d = 1; d < 16; d <<= 1) l += __shfl_xor(l, d, 16);
        run_l[i] = run_l[i] * scale + l;
        run_m[i] = nm;
#pragma unroll
        for (int j = 0; j < 4; ++j) O[j][i] *= scale;
        int r = hw * 8 + i;
        Pw[r * 32 + ln]      = (bf16_t)p0;
        Pw[r * 32 + 16 + ln] = (bf16_t)p1;
      }
      // same-wave cross-lane LDS RAW: force compile-time order + hw completion
      __asm__ __volatile__("s_wait_dscnt 0" ::: "memory");

      // ---- P (16x32) x V (32x64): one WMMA per dk tile
      FragB pa;
      {
        const uint4* pp = reinterpret_cast<const uint4*>(Pw + ln * 32);
        pa.q[0] = pp[hw ? 1 : 0];
        pa.q[1] = pp[hw ? 3 : 2];
      }
#pragma unroll
      for (int j = 0; j < 4; ++j) {
        FragB vb;
        const uint4* vp = reinterpret_cast<const uint4*>(&vT[(j * 16 + ln) * 32]);
        vb.q[0] = vp[hw ? 1 : 0];
        vb.q[1] = vp[hw ? 3 : 2];
        O[j] = wmma_bf16(pa.v, vb.v, O[j]);
      }
    }
    __syncthreads();
  }

  // normalize + store head-concat layout [token, HID]
  int b = bh >> 4, h = bh & 15;
#pragma unroll
  for (int i = 0; i < 8; ++i) {
    int s = s0w + hw * 8 + i;
    if (s < SEQ) {
      float inv = 1.0f / run_l[i];
      long row = (long)(b * SEQ + s) * HID + h * DK;
#pragma unroll
      for (int j = 0; j < 4; ++j)
        attnb[row + j * 16 + ln] = (bf16_t)(O[j][i] * inv);
    }
  }
}

// ---------------- 4) residual + layernorm ----------------
__global__ __launch_bounds__(256) void ln_kernel(const float* __restrict__ xres,
                                                 const float* __restrict__ delta,
                                                 const float* __restrict__ g,
                                                 const float* __restrict__ bta,
                                                 float* __restrict__ outf,
                                                 bf16_t* __restrict__ outb) {
  __shared__ float red[256];
  int m = blockIdx.x;
  int tid = threadIdx.x;
  float v[4];
#pragma unroll
  for (int u = 0; u < 4; ++u) {
    int c = tid * 4 + u;
    v[u] = xres[(long)m * HID + c] + delta[(long)m * HID + c];
  }
  float s = v[0] + v[1] + v[2] + v[3];
  red[tid] = s;
  __syncthreads();
  for (int off = 128; off > 0; off >>= 1) {
    if (tid < off) red[tid] += red[tid + off];
    __syncthreads();
  }
  float mu = red[0] * (1.0f / HID);
  __syncthreads();
  float vs = 0.f;
#pragma unroll
  for (int u = 0; u < 4; ++u) { float d = v[u] - mu; vs += d * d; }
  red[tid] = vs;
  __syncthreads();
  for (int off = 128; off > 0; off >>= 1) {
    if (tid < off) red[tid] += red[tid + off];
    __syncthreads();
  }
  float var = red[0] * (1.0f / HID);
  float rinv = rsqrtf(var + 1e-5f);
#pragma unroll
  for (int u = 0; u < 4; ++u) {
    int c = tid * 4 + u;
    float o = (v[u] - mu) * rinv * g[c] + bta[c];
    if (outf) outf[(long)m * HID + c] = o;
    outb[(long)m * HID + c] = (bf16_t)o;
  }
}

// ---------------- launch ----------------
extern "C" void kernel_launch(void* const* d_in, const int* in_sizes, int n_in,
                              void* d_out, int out_size, void* d_ws, size_t ws_size,
                              hipStream_t stream) {
  (void)in_sizes; (void)n_in; (void)out_size;
  const int*   tokens = (const int*)  d_in[0];
  const float* emb    = (const float*)d_in[1];
  const float* Wq     = (const float*)d_in[2];
  const float* bq     = (const float*)d_in[3];
  const float* Wk     = (const float*)d_in[4];
  const float* bk     = (const float*)d_in[5];
  const float* Wv     = (const float*)d_in[6];
  const float* bv     = (const float*)d_in[7];
  const float* Wc     = (const float*)d_in[8];
  const float* bc     = (const float*)d_in[9];
  const float* ln1g   = (const float*)d_in[10];
  const float* ln1b   = (const float*)d_in[11];
  const float* W1     = (const float*)d_in[12];
  const float* b1     = (const float*)d_in[13];
  const float* W2     = (const float*)d_in[14];
  const float* b2     = (const float*)d_in[15];
  const float* ln2g   = (const float*)d_in[16];
  const float* ln2b   = (const float*)d_in[17];
  const float* Wlog   = (const float*)d_in[18];
  const float* blog   = (const float*)d_in[19];

  char* ws = (char*)d_ws;
  size_t off = 0;
  auto alloc = [&](size_t bytes) -> char* {
    char* p = ws + off;
    off += (bytes + 255) & ~(size_t)255;
    return p;
  };
  float*  xf    = (float*) alloc((size_t)MPAD * HID * 4);
  bf16_t* xb    = (bf16_t*)alloc((size_t)MPAD * HID * 2);
  bf16_t* q     = (bf16_t*)alloc((size_t)BATCH * NHEAD * SPAD * DK * 2);
  bf16_t* k     = (bf16_t*)alloc((size_t)BATCH * NHEAD * SPAD * DK * 2);
  bf16_t* v     = (bf16_t*)alloc((size_t)BATCH * NHEAD * SPAD * DK * 2);
  bf16_t* attn  = (bf16_t*)alloc((size_t)MPAD * HID * 2);
  float*  cond  = (float*) alloc((size_t)MPAD * HID * 4);
  float*  x1f   = (float*) alloc((size_t)MPAD * HID * 4);
  bf16_t* x1b   = (bf16_t*)alloc((size_t)MPAD * HID * 2);
  bf16_t* hb    = (bf16_t*)alloc((size_t)MPAD * FFH * 2);
  float*  ff    = (float*) alloc((size_t)MPAD * HID * 4);
  bf16_t* x2b   = (bf16_t*)alloc((size_t)MPAD * HID * 2);
  bf16_t* WqT   = (bf16_t*)alloc((size_t)HID * HID * 2);
  bf16_t* WkT   = (bf16_t*)alloc((size_t)HID * HID * 2);
  bf16_t* WvT   = (bf16_t*)alloc((size_t)HID * HID * 2);
  bf16_t* WcT   = (bf16_t*)alloc((size_t)HID * HID * 2);
  bf16_t* W1T   = (bf16_t*)alloc((size_t)FFH * HID * 2);
  bf16_t* W2T   = (bf16_t*)alloc((size_t)HID * FFH * 2);
  bf16_t* WlogT = (bf16_t*)alloc((size_t)VOCAB * HID * 2);
  if (off > ws_size) return;  // insufficient scratch: bail deterministically

  // weight prep: bf16, transposed [N][K] (whole set ~80 MB -> L2 resident)
  prep_kernel<<<dim3(16, 16),  256, 0, stream>>>(Wq,   WqT,   HID, 65536, 64);
  prep_kernel<<<dim3(16, 16),  256, 0, stream>>>(Wk,   WkT,   HID, 65536, 64);
  prep_kernel<<<dim3(16, 16),  256, 0, stream>>>(Wv,   WvT,   HID, 65536, 64);
  prep_kernel<<<dim3(16, 16),  256, 0, stream>>>(Wc,   WcT,   HID, 64, HID);
  prep_kernel<<<dim3(32, 16),  256, 0, stream>>>(W1,   W1T,   HID, 64, FFH);
  prep_kernel<<<dim3(16, 32),  256, 0, stream>>>(W2,   W2T,   FFH, 64, HID);
  prep_kernel<<<dim3(500, 16), 256, 0, stream>>>(Wlog, WlogT, HID, 64, VOCAB);

  embed_kernel<<<MPAD, 256, 0, stream>>>(tokens, emb, xf, xb);

  dim3 blk(128);
  gemm_kernel<<<dim3(16, 16), blk, 0, stream>>>(xb, WqT, bq, q, HID, HID, 1, 0);
  gemm_kernel<<<dim3(16, 16), blk, 0, stream>>>(xb, WkT, bk, k, HID, HID, 1, 0);
  gemm_kernel<<<dim3(16, 16), blk, 0, stream>>>(xb, WvT, bv, v, HID, HID, 1, 0);

  attn_kernel<<<dim3(8, 64), 256, 0, stream>>>(q, k, v, attn);

  gemm_kernel<<<dim3(16, 16), blk, 0, stream>>>(attn, WcT, bc, cond, HID, HID, 2, 0);
  ln_kernel<<<MVALID, 256, 0, stream>>>(xf, cond, ln1g, ln1b, x1f, x1b);

  gemm_kernel<<<dim3(32, 16), blk, 0, stream>>>(x1b, W1T, b1, hb, FFH, HID, 0, 1);
  gemm_kernel<<<dim3(16, 16), blk, 0, stream>>>(hb, W2T, b2, ff, HID, FFH, 2, 0);
  ln_kernel<<<MVALID, 256, 0, stream>>>(x1f, ff, ln2g, ln2b, nullptr, x2b);

  gemm_kernel<<<dim3(500, 16), blk, 0, stream>>>(x2b, WlogT, blog, (float*)d_out,
                                                 VOCAB, HID, 2, 0);
}
